// EGatConv_67388036874511
// MI455X (gfx1250) — compile-verified, matrix operand order checked by reading, not run on hardware
//
#include <hip/hip_runtime.h>
#include <hip/hip_bf16.h>
#include <math.h>

// ---------------- problem constants (match reference) ----------------
#define NN 50000      // nodes
#define NE 1600000    // edges
#define DIN 128       // node_in_dim
#define DD 64         // node_out_dim / GRU hidden
#define DED 3         // edge_in_dim
#define NSTEPS 3
#define BNEPS 1e-5f
#define LSLOPE 0.2f

typedef __attribute__((ext_vector_type(16))) __bf16 v16bf;
typedef __attribute__((ext_vector_type(8)))  float  v8f;

// ---------------- helpers ----------------
__device__ __forceinline__ unsigned int fkey(float f) {
    unsigned int b = __float_as_uint(f);
    return (b & 0x80000000u) ? ~b : (b | 0x80000000u);
}
__device__ __forceinline__ float fdecode(unsigned int k) {
    unsigned int b = (k & 0x80000000u) ? (k & 0x7FFFFFFFu) : ~k;
    return __uint_as_float(b);
}
__device__ __forceinline__ float sigm(float x) { return 1.f / (1.f + expf(-x)); }

// ---------------- B pre-pack: fp32 weight -> bf16 WMMA fragment layout ----------------
// Fragment storage: [tile_n][k_chunk][lane][16 bf16]  (32B per lane -> one aligned vec load)
// transposed=0: source is B row-major (K x Ncol), B[k*Ncol + n]
// transposed=1: source is W row-major (Ncol x K), B[k][n] = W[n*K + k]   (for w_ih/w_hh^T)
__global__ void pack_b_bf16(const float* __restrict__ src, __bf16* __restrict__ out,
                            int K, int Ncol, int transposed) {
    const int kch = K >> 5;
    const int total = (Ncol >> 4) * kch * 32;
    int idx = blockIdx.x * blockDim.x + threadIdx.x;
    if (idx >= total) return;
    const int lane = idx & 31;
    const int rest = idx >> 5;
    const int kc = rest % kch;
    const int tn = rest / kch;
    const int n  = tn * 16 + (lane & 15);
    const int kb = kc * 32 + (((lane >> 4) & 1) << 3);
    __bf16* o = out + (size_t)idx * 16;
    #pragma unroll
    for (int e = 0; e < 16; ++e) {
        const int k = kb + ((e < 8) ? e : (8 + e));   // ISA 16-bit fragment k pattern
        const float v = transposed ? src[(size_t)n * K + k] : src[(size_t)k * Ncol + n];
        o[e] = (__bf16)v;
    }
}

// ---------------- WMMA GEMM: C[M x NCOL] = A[M x K] @ B[K x NCOL] ----------------
// Whole packed-B (<=24KB bf16) is staged in LDS once per block and shared by all
// 8 waves; the k-loop then pairs streaming A (global, fp32->bf16 in-register)
// with ds_load_b128 B fragments. One wave computes a 16x64 strip (4 n-tiles);
// K/NCOL compile-time -> fully unrolled, KCH*4 static v_wmma per instantiation.
template<int K, int NCOL, bool HAS_BIAS, bool HAS_ADD>
__global__ void wmma_gemm_strip(const float* __restrict__ A,
                                const __bf16* __restrict__ Bp,
                                float* __restrict__ C,
                                const float* __restrict__ bias,
                                const float* __restrict__ addsrc,
                                int M) {
    constexpr int KCH = K / 32;
    constexpr int STRIPS_N = NCOL / 64;                    // 64-wide strips per row
    constexpr int BELEMS = K * NCOL;                       // packed B element count

    __shared__ __bf16 bs[BELEMS];

    // cooperative stage of entire packed B into LDS (uint4 = 16B chunks)
    {
        const uint4* g = (const uint4*)Bp;
        uint4* l = (uint4*)bs;
        #pragma unroll
        for (int i = threadIdx.x; i < BELEMS / 8; i += 256) l[i] = g[i];
    }
    __syncthreads();

    const int wave = threadIdx.x >> 5;
    const int lane = threadIdx.x & 31;
    const int strip = blockIdx.x * 8 + wave;
    if (strip >= (M >> 4) * STRIPS_N) return;              // wave-uniform exit
    const int tm = strip / STRIPS_N;
    const int sn = strip - tm * STRIPS_N;

    const int l15 = lane & 15;
    const int kb  = ((lane >> 4) & 1) << 3;                // 0 or 8 (ISA 16-bit layout)
    const float* __restrict__ arow = A + (size_t)(tm * 16 + l15) * K;
    // this wave's strip base inside LDS-staged packed B
    const __bf16* bbase = bs + (((size_t)(sn * 4) * KCH) * 32 + (size_t)lane) * 16;
    constexpr size_t TILE_STRIDE = (size_t)KCH * 32 * 16;  // elems between n-tiles

    v8f acc[4] = {};
    #pragma unroll
    for (int kc = 0; kc < KCH; ++kc) {
        const int k0 = kc * 32;
        __builtin_prefetch(arow + k0 + 32, 0, 1);          // global_prefetch_b8

        // A fragment: lane holds row m; k = k0+kb+e (e<8) and k0+kb+8+e (e>=8)
        const float4 a0 = *(const float4*)(arow + k0 + kb + 0);
        const float4 a1 = *(const float4*)(arow + k0 + kb + 4);
        const float4 a2 = *(const float4*)(arow + k0 + kb + 16);
        const float4 a3 = *(const float4*)(arow + k0 + kb + 20);
        v16bf af;
        af[0]=(__bf16)a0.x;  af[1]=(__bf16)a0.y;  af[2]=(__bf16)a0.z;  af[3]=(__bf16)a0.w;
        af[4]=(__bf16)a1.x;  af[5]=(__bf16)a1.y;  af[6]=(__bf16)a1.z;  af[7]=(__bf16)a1.w;
        af[8]=(__bf16)a2.x;  af[9]=(__bf16)a2.y;  af[10]=(__bf16)a2.z; af[11]=(__bf16)a2.w;
        af[12]=(__bf16)a3.x; af[13]=(__bf16)a3.y; af[14]=(__bf16)a3.z; af[15]=(__bf16)a3.w;

        const __bf16* bkc = bbase + (size_t)kc * 32 * 16;
        #pragma unroll
        for (int j = 0; j < 4; ++j) {
            const v16bf bf = *(const v16bf*)(bkc + (size_t)j * TILE_STRIDE);  // ds_load x2
            acc[j] = __builtin_amdgcn_wmma_f32_16x16x32_bf16(
                false, af, false, bf, (short)0, acc[j], false, false);
        }
    }

    // epilogue: C[m][n] = acc + bias[n] (+ addsrc[m][n])
    const int mb = tm * 16 + kb;                           // row base for lane's C elems
    #pragma unroll
    for (int j = 0; j < 4; ++j) {
        const int ncol = (sn * 4 + j) * 16 + l15;
        const float bv = HAS_BIAS ? bias[ncol] : 0.f;
        #pragma unroll
        for (int v = 0; v < 8; ++v) {
            const size_t idx = (size_t)(mb + v) * NCOL + ncol;
            float r = acc[j][v] + bv;
            if (HAS_ADD) r += addsrc[idx];
            C[idx] = r;
        }
    }
}

// ---------------- small utility kernels ----------------
__global__ void zero_f32(float* p, long long n) {
    long long i = (long long)blockIdx.x * blockDim.x + threadIdx.x;
    if (i < n) p[i] = 0.f;
}

// BatchNorm statistics: per-block LDS reduction, 128 atomics per block
__global__ void bn_stats(const float* __restrict__ h0, float* __restrict__ sums, int n) {
    __shared__ float sh_s[256], sh_q[256];
    const int t = threadIdx.x;
    const int col = t & 63;
    const int rl = t >> 6;                    // 0..3
    float s = 0.f, q = 0.f;
    const int row0 = blockIdx.x * 256;
    for (int i = 0; i < 64; ++i) {
        int row = row0 + rl * 64 + i;
        if (row < n) {
            float v = h0[(size_t)row * DD + col];
            s += v; q += v * v;
        }
    }
    sh_s[t] = s; sh_q[t] = q;
    __syncthreads();
    if (t < 64) {
        float S = sh_s[t] + sh_s[t + 64] + sh_s[t + 128] + sh_s[t + 192];
        float Q = sh_q[t] + sh_q[t + 64] + sh_q[t + 128] + sh_q[t + 192];
        atomicAdd(&sums[col], S);
        atomicAdd(&sums[64 + col], Q);
    }
}

__global__ void bn_apply_relu(const float* __restrict__ h0, const float* __restrict__ sums,
                              const float* __restrict__ g, const float* __restrict__ b,
                              float* __restrict__ out, int total) {
    int i = blockIdx.x * blockDim.x + threadIdx.x;
    if (i >= total) return;
    int c = i & 63;
    const float inv_n = 1.f / (float)NN;
    float mu  = sums[c] * inv_n;
    float var = sums[64 + c] * inv_n - mu * mu;
    float y = (h0[i] - mu) * rsqrtf(var + BNEPS) * g[c] + b[c];
    out[i] = fmaxf(y, 0.f);
}

// per-node attention halves: a_dst[n] = xt[n].att_dst ; a_src[n] = xt[n].att_src
__global__ void node_dots(const float* __restrict__ xt,
                          const float* __restrict__ att_dst, const float* __restrict__ att_src,
                          float* __restrict__ a_dst, float* __restrict__ a_src, int n) {
    int i = blockIdx.x * blockDim.x + threadIdx.x;
    if (i >= n) return;
    const float* row = xt + (size_t)i * DD;
    float ad = 0.f, as = 0.f;
    #pragma unroll
    for (int d = 0; d < DD; ++d) {
        float v = row[d];
        ad += v * att_dst[d];
        as += v * att_src[d];
    }
    a_dst[i] = ad; a_src[i] = as;
}

// per-edge logits + LeakyReLU + segment-max (monotone uint key atomicMax)
__global__ void edge_logits(const int* __restrict__ src, const int* __restrict__ dst,
                            const float* __restrict__ ea, const float* __restrict__ att_e,
                            const float* __restrict__ a_src, const float* __restrict__ a_dst,
                            float* __restrict__ elog, unsigned int* __restrict__ segmax, int e_cnt) {
    int e = blockIdx.x * blockDim.x + threadIdx.x;
    if (e >= e_cnt) return;
    int s = src[e], d = dst[e];
    float l = a_dst[d] + a_src[s]
            + ea[(size_t)e * DED + 0] * att_e[0]
            + ea[(size_t)e * DED + 1] * att_e[1]
            + ea[(size_t)e * DED + 2] * att_e[2];
    l = (l >= 0.f) ? l : LSLOPE * l;
    elog[e] = l;
    atomicMax(&segmax[d], fkey(l));
}

// exp(logit - segmax[dst]) + segment-sum denominator
__global__ void edge_exp(const int* __restrict__ dst,
                         const unsigned int* __restrict__ segmax,
                         float* __restrict__ elog, float* __restrict__ denom, int e_cnt) {
    int e = blockIdx.x * blockDim.x + threadIdx.x;
    if (e >= e_cnt) return;
    int d = dst[e];
    unsigned int k = segmax[d];
    float smax = (k == 0u) ? 0.f : fdecode(k);
    float ex = expf(elog[e] - smax);
    elog[e] = ex;
    atomicAdd(&denom[d], ex);
}

// attention-weighted scatter-add: agg[dst] += (exp/denom[dst]) * xt[src]
// one thread per (edge, 2-element chunk) -> E*32 threads
__global__ void edge_aggregate(const int* __restrict__ src, const int* __restrict__ dst,
                               const float* __restrict__ elog, const float* __restrict__ denom,
                               const float* __restrict__ xt, float* __restrict__ agg,
                               long long total) {
    long long tid = (long long)blockIdx.x * blockDim.x + threadIdx.x;
    if (tid >= total) return;
    int e = (int)(tid >> 5);
    int c = (int)(tid & 31) << 1;
    int s = src[e], d = dst[e];
    float wgt = elog[e] / denom[d];
    float2 v = *(const float2*)(xt + (size_t)s * DD + c);
    atomicAdd(&agg[(size_t)d * DD + c],     wgt * v.x);
    atomicAdd(&agg[(size_t)d * DD + c + 1], wgt * v.y);
}

__global__ void relu_inplace(float* p, int total) {
    int i = blockIdx.x * blockDim.x + threadIdx.x;
    if (i < total) p[i] = fmaxf(p[i], 0.f);
}

// GRU cell gates, h (=out) updated in place; gi/gh are [N x 192]
__global__ void gru_update(const float* __restrict__ gi, const float* __restrict__ gh,
                           float* __restrict__ h, int total) {
    int i = blockIdx.x * blockDim.x + threadIdx.x;
    if (i >= total) return;
    int n = i >> 6, d = i & 63;
    const float* gin = gi + (size_t)n * (3 * DD);
    const float* ghn = gh + (size_t)n * (3 * DD);
    float r  = sigm(gin[d]            + ghn[d]);
    float z  = sigm(gin[DD + d]       + ghn[DD + d]);
    float nn = tanhf(gin[2 * DD + d] + r * ghn[2 * DD + d]);
    h[i] = (1.f - z) * nn + z * h[i];
}

// ---------------- host orchestration ----------------
static inline unsigned int cdiv_u(long long a, long long b) { return (unsigned int)((a + b - 1) / b); }

extern "C" void kernel_launch(void* const* d_in, const int* in_sizes, int n_in,
                              void* d_out, int out_size, void* d_ws, size_t ws_size,
                              hipStream_t stream) {
    (void)in_sizes; (void)n_in; (void)out_size; (void)ws_size;

    const float* x        = (const float*)d_in[0];
    const float* edge_at  = (const float*)d_in[1];
    const float* mlp_w    = (const float*)d_in[2];
    const float* mlp_b    = (const float*)d_in[3];
    const float* bn_g     = (const float*)d_in[4];
    const float* bn_b     = (const float*)d_in[5];
    const float* conv_w   = (const float*)d_in[6];
    const float* att_dst  = (const float*)d_in[7];
    const float* att_src  = (const float*)d_in[8];
    const float* att_edge = (const float*)d_in[9];
    const float* w_ih     = (const float*)d_in[10];
    const float* w_hh     = (const float*)d_in[11];
    const float* b_ih     = (const float*)d_in[12];
    const float* b_hh     = (const float*)d_in[13];
    const float* lin_w    = (const float*)d_in[14];
    const float* lin_b    = (const float*)d_in[15];
    const int*   eidx     = (const int*)d_in[16];
    const int* src = eidx;
    const int* dst = eidx + NE;

    // ---- workspace carve (256B aligned) ----
    char* wsp = (char*)d_ws;
    auto carve = [&](size_t bytes) -> void* {
        void* p = (void*)wsp;
        wsp += (bytes + 255) & ~(size_t)255;
        return p;
    };
    float* h0     = (float*)carve((size_t)NN * DD * 4);
    float* outbuf = (float*)carve((size_t)NN * DD * 4);   // out == GRU hidden h
    float* xt     = (float*)carve((size_t)NN * DD * 4);
    float* mbuf   = (float*)carve((size_t)NN * DD * 4);   // agg, then relu'd message
    float* gi     = (float*)carve((size_t)NN * 3 * DD * 4);
    float* gh     = (float*)carve((size_t)NN * 3 * DD * 4);
    float* elog   = (float*)carve((size_t)NE * 4);
    float* a_dst  = (float*)carve((size_t)NN * 4);
    float* a_src  = (float*)carve((size_t)NN * 4);
    float* denom  = (float*)carve((size_t)NN * 4);
    unsigned int* segmax = (unsigned int*)carve((size_t)NN * 4);
    float* sums   = (float*)carve(128 * 4);               // [0:64] sum, [64:128] sumsq
    __bf16* mlp_p = (__bf16*)carve((size_t)DIN * DD * 2);       // packed bf16 fragments
    __bf16* conv_p= (__bf16*)carve((size_t)DD * DD * 2);
    __bf16* ih_p  = (__bf16*)carve((size_t)DD * 3 * DD * 2);
    __bf16* hh_p  = (__bf16*)carve((size_t)DD * 3 * DD * 2);
    __bf16* lin_p = (__bf16*)carve((size_t)DIN * DD * 2);

    const dim3 B256(256);

    // ---- weight packing (tiny) ----
    pack_b_bf16<<<cdiv_u((DD/16) * (DIN/32) * 32, 256), B256, 0, stream>>>(mlp_w,  mlp_p,  DIN, DD,     0);
    pack_b_bf16<<<cdiv_u((DD/16) * (DD/32)  * 32, 256), B256, 0, stream>>>(conv_w, conv_p, DD,  DD,     0);
    pack_b_bf16<<<cdiv_u((3*DD/16) * (DD/32) * 32, 256), B256, 0, stream>>>(w_ih,  ih_p,   DD,  3*DD,   1);
    pack_b_bf16<<<cdiv_u((3*DD/16) * (DD/32) * 32, 256), B256, 0, stream>>>(w_hh,  hh_p,   DD,  3*DD,   1);
    pack_b_bf16<<<cdiv_u((DD/16) * (DIN/32) * 32, 256), B256, 0, stream>>>(lin_w,  lin_p,  DIN, DD,     0);

    // h0 = x @ mlp_w + mlp_b    (50000x128x64)
    {
        unsigned int strips = (NN / 16) * (DD / 64);
        wmma_gemm_strip<DIN, DD, true, false><<<cdiv_u(strips, 8), B256, 0, stream>>>(
            x, mlp_p, h0, mlp_b, nullptr, NN);
    }
    // BatchNorm (batch stats) + ReLU -> outbuf
    zero_f32<<<1, 128, 0, stream>>>(sums, 128);
    bn_stats<<<cdiv_u(NN, 256), B256, 0, stream>>>(h0, sums, NN);
    bn_apply_relu<<<cdiv_u((long long)NN * DD, 256), B256, 0, stream>>>(h0, sums, bn_g, bn_b, outbuf, NN * DD);

    for (int step = 0; step < NSTEPS; ++step) {
        // xt = out @ conv_w      (50000x64x64)
        {
            unsigned int strips = (NN / 16) * (DD / 64);
            wmma_gemm_strip<DD, DD, false, false><<<cdiv_u(strips, 8), B256, 0, stream>>>(
                outbuf, conv_p, xt, nullptr, nullptr, NN);
        }
        node_dots<<<cdiv_u(NN, 256), B256, 0, stream>>>(xt, att_dst, att_src, a_dst, a_src, NN);

        // reset segment buffers
        zero_f32<<<cdiv_u(NN, 256), B256, 0, stream>>>(denom, NN);
        zero_f32<<<cdiv_u(NN, 256), B256, 0, stream>>>((float*)segmax, NN);  // key 0 == -inf
        zero_f32<<<cdiv_u((long long)NN * DD, 256), B256, 0, stream>>>(mbuf, (long long)NN * DD);

        edge_logits<<<cdiv_u(NE, 256), B256, 0, stream>>>(src, dst, edge_at, att_edge,
                                                          a_src, a_dst, elog, segmax, NE);
        edge_exp<<<cdiv_u(NE, 256), B256, 0, stream>>>(dst, segmax, elog, denom, NE);
        edge_aggregate<<<cdiv_u((long long)NE * 32, 256), B256, 0, stream>>>(
            src, dst, elog, denom, xt, mbuf, (long long)NE * 32);
        relu_inplace<<<cdiv_u((long long)NN * DD, 256), B256, 0, stream>>>(mbuf, NN * DD);

        // GRU: gi = m @ w_ih^T + b_ih ; gh = h @ w_hh^T + b_hh  (50000x64x192 each)
        {
            unsigned int strips = (NN / 16) * ((3 * DD) / 64);
            wmma_gemm_strip<DD, 3 * DD, true, false><<<cdiv_u(strips, 8), B256, 0, stream>>>(
                mbuf, ih_p, gi, b_ih, nullptr, NN);
            wmma_gemm_strip<DD, 3 * DD, true, false><<<cdiv_u(strips, 8), B256, 0, stream>>>(
                outbuf, hh_p, gh, b_hh, nullptr, NN);
        }
        gru_update<<<cdiv_u((long long)NN * DD, 256), B256, 0, stream>>>(gi, gh, outbuf, NN * DD);
    }

    // y = x @ lin_w + lin_b + out   (fused epilogue into d_out)
    {
        unsigned int strips = (NN / 16) * (DD / 64);
        wmma_gemm_strip<DIN, DD, true, true><<<cdiv_u(strips, 8), B256, 0, stream>>>(
            x, lin_p, (float*)d_out, lin_b, outbuf, NN);
    }
}
